// MultiHeadSelfAttentionBlock_75746043232658
// MI455X (gfx1250) — matrix operand order repaired and checked by reading, not compiled, not run on hardware
//
#include <hip/hip_runtime.h>

// MI455X / gfx1250 multi-head self-attention block.
// All GEMMs via v_wmma_f32_16x16x32_f16 (f16 inputs, f32 accumulate).
// Weights pre-converted to f16 (native B layout) to kill inner-loop cvts
// and WMMA->VALU WAR hazard NOPs. 32x64 output tile per wave in the
// projection GEMMs. Flash-style online-softmax attention per 16-query tile.

typedef __attribute__((ext_vector_type(16))) _Float16 v16h;
typedef __attribute__((ext_vector_type(8)))  _Float16 v8h;
typedef __attribute__((ext_vector_type(8)))  float    v8f;
typedef __attribute__((ext_vector_type(4)))  float    v4f;

#define DMODEL 1024
#define SEQ    2048
#define BATCH  2
#define HEADS  16
#define DKH    64
#define HEADELEMS ((size_t)BATCH * HEADS * SEQ * DKH)     // 4 M f16 = 8 MB
#define WELEMS    ((size_t)DMODEL * DMODEL)               // 1 M f16 = 2 MB

__device__ __forceinline__ v8f wmma_f16(v16h a, v16h b, v8f c) {
  return __builtin_amdgcn_wmma_f32_16x16x32_f16(false, a, false, b,
                                                (short)0, c, false, false);
}

// A-matrix (16x32 f16) from fp32 row-major: lane holds row (lane&15);
// elems 0..15 -> K = kc..kc+15, kc=(lane>>4)*16.
__device__ __forceinline__ v16h load_a_f32(const float* __restrict__ rowk, int kc) {
  const v4f* p0 = (const v4f*)(rowk + kc);
  const v4f* p1 = (const v4f*)(rowk + kc + 8);
  v4f f0 = p0[0], f1 = p0[1], f2 = p1[0], f3 = p1[1];
  v16h a;
#pragma unroll
  for (int i = 0; i < 4; ++i) {
    a[i]      = (_Float16)f0[i];
    a[4 + i]  = (_Float16)f1[i];
    a[8 + i]  = (_Float16)f2[i];
    a[12 + i] = (_Float16)f3[i];
  }
  return a;
}

// A-matrix (16x32 f16) from f16 row-major.
__device__ __forceinline__ v16h load_a_f16(const _Float16* __restrict__ rowk, int kc) {
  v8h c0 = *(const v8h*)(rowk + kc);
  v8h c1 = *(const v8h*)(rowk + kc + 8);
  v16h a;
#pragma unroll
  for (int i = 0; i < 8; ++i) { a[i] = c0[i]; a[8 + i] = c1[i]; }
  return a;
}

// ---------------- Kernel 0: fp32 -> f16 weight conversion ----------------
// Layout preserved [N][K]: already the native WMMA B layout for x @ W^T.
__global__ __launch_bounds__(256)
void convert_weights(const float* __restrict__ wq, const float* __restrict__ wk,
                     const float* __restrict__ wv, const float* __restrict__ wo,
                     _Float16* __restrict__ dq, _Float16* __restrict__ dk,
                     _Float16* __restrict__ dv, _Float16* __restrict__ dwo) {
  const float* src; _Float16* dst;
  switch (blockIdx.z) {
    case 0:  src = wq; dst = dq;  break;
    case 1:  src = wk; dst = dk;  break;
    case 2:  src = wv; dst = dv;  break;
    default: src = wo; dst = dwo; break;
  }
  const size_t i = ((size_t)blockIdx.x * blockDim.x + threadIdx.x) * 8;
  v4f f0 = *(const v4f*)(src + i);
  v4f f1 = *(const v4f*)(src + i + 4);
  v8h o;
#pragma unroll
  for (int j = 0; j < 4; ++j) { o[j] = (_Float16)f0[j]; o[4 + j] = (_Float16)f1[j]; }
  *(v8h*)(dst + i) = o;
}

// ---------------- Kernel 1: fused Q/K/V projections (z = 0/1/2) ----------------
// Y = X @ W^T + b in f16; 32 rows x 64 cols per wave (8 WMMAs / k-step).
// Q,K stored [B,H,S,DK]; V stored transposed [B,H,DK,S].
__global__ __launch_bounds__(256)
void proj_qkv(const float* __restrict__ q, const float* __restrict__ k,
              const float* __restrict__ v,
              const _Float16* __restrict__ wqh, const _Float16* __restrict__ wkh,
              const _Float16* __restrict__ wvh,
              const float* __restrict__ bq, const float* __restrict__ bk,
              const float* __restrict__ bv,
              _Float16* __restrict__ Qh, _Float16* __restrict__ Kh,
              _Float16* __restrict__ Vt) {
  const int lane = threadIdx.x & 31;
  const int wid  = threadIdx.x >> 5;
  const int task = blockIdx.x * 8 + wid;        // 2048 wave-tasks
  const int mtile = task >> 4;                  // 0..127 (32-row tile of B*S)
  const int ncb   = (task & 15) * 64;           // 64-col group

  const float* x; const _Float16* wh; const float* bias;
  const int z = blockIdx.z;
  if (z == 0)      { x = q; wh = wqh; bias = bq; }
  else if (z == 1) { x = k; wh = wkh; bias = bk; }
  else             { x = v; wh = wvh; bias = bv; }

  const int kc      = (lane >> 4) * 16;
  const int khalf16 = (lane >> 4) * 16;
  const int hi8     = (lane >> 4) * 8;
  const int gm0     = mtile * 32 + (lane & 15);
  const float* arow0 = x + (size_t)gm0 * DMODEL;
  const float* arow1 = arow0 + (size_t)16 * DMODEL;
  const int ncol0   = ncb + (lane & 15);

  v8f acc0[4], acc1[4];
#pragma unroll
  for (int t = 0; t < 4; ++t) { acc0[t] = (v8f){}; acc1[t] = (v8f){}; }

  for (int kk = 0; kk < DMODEL; kk += 32) {
    v16h a0 = load_a_f32(arow0 + kk, kc);
    v16h a1 = load_a_f32(arow1 + kk, kc);
#pragma unroll
    for (int t = 0; t < 4; ++t) {
      v16h b = *(const v16h*)(wh + (size_t)(ncol0 + t * 16) * DMODEL + kk + khalf16);
      acc0[t] = wmma_f16(a0, b, acc0[t]);
      acc1[t] = wmma_f16(a1, b, acc1[t]);
    }
  }

#pragma unroll
  for (int t = 0; t < 4; ++t) {
    const int col = ncol0 + t * 16;
    const float bsv = bias[col];
    const int hh = col >> 6, dk = col & 63;
#pragma unroll
    for (int rt = 0; rt < 2; ++rt) {
#pragma unroll
      for (int r = 0; r < 8; ++r) {
        const int g = mtile * 32 + rt * 16 + r + hi8;
        const int bidx = g >> 11, sidx = g & (SEQ - 1);
        const float val = (rt ? acc1[t][r] : acc0[t][r]) + bsv;
        if (z == 2)
          Vt[(((size_t)bidx * HEADS + hh) * DKH + dk) * SEQ + sidx] = (_Float16)val;
        else {
          _Float16* dst = (z == 0) ? Qh : Kh;
          dst[(((size_t)bidx * HEADS + hh) * SEQ + sidx) * DKH + dk] = (_Float16)val;
        }
      }
    }
  }
}

// ---------------- Kernel 2: flash attention per 16-query tile ----------------
__global__ __launch_bounds__(256)
void attn_flash(const _Float16* __restrict__ Qh, const _Float16* __restrict__ Kh,
                const _Float16* __restrict__ Vt, const int* __restrict__ mask,
                _Float16* __restrict__ Ctx) {
  __shared__ _Float16 plds[8 * 512];            // 16x32 f16 P-tile per wave
  const int lane = threadIdx.x & 31;
  const int wid  = threadIdx.x >> 5;
  const int task = blockIdx.x * 8 + wid;        // 4096 = B*H*(S/16)
  const int qt = task & 127;
  const int h  = (task >> 7) & 15;
  const int b  = task >> 11;

  const int n       = lane & 15;
  const int hi8     = (lane >> 4) * 8;
  const int kc      = (lane >> 4) * 16;
  const int khalf16 = (lane >> 4) * 16;
  _Float16* myl = plds + wid * 512;

  const _Float16* qrowp =
      Qh + (((size_t)b * HEADS + h) * SEQ + qt * 16 + (lane & 15)) * DKH;
  const v16h aq0 = load_a_f16(qrowp, kc);        // dk 0..31
  const v16h aq1 = load_a_f16(qrowp + 32, kc);   // dk 32..63

  const _Float16* kbaseH = Kh + ((size_t)b * HEADS + h) * SEQ * DKH;
  const _Float16* vbaseH = Vt + ((size_t)b * HEADS + h) * DKH * SEQ;

  float mrow[8], lrow[8];
  v8f acc[4];
#pragma unroll
  for (int r = 0; r < 8; ++r) { mrow[r] = -3.0e38f; lrow[r] = 0.0f; }
#pragma unroll
  for (int t = 0; t < 4; ++t) acc[t] = (v8f){};

  const int qrow0 = qt * 16;

  for (int kt = 0; kt < SEQ / 32; ++kt) {
    const int key0 = kt * 32;
    // ---- scores S = Q K^T (two 16-key halves, K-dim 64 = 2 WMMAs each) ----
    const _Float16* kb0 = kbaseH + (size_t)(key0 + n) * DKH + khalf16;
    const _Float16* kb1 = kbaseH + (size_t)(key0 + 16 + n) * DKH + khalf16;
    v16h bl0 = *(const v16h*)kb0;
    v16h bh0 = *(const v16h*)(kb0 + 32);
    v16h bl1 = *(const v16h*)kb1;
    v16h bh1 = *(const v16h*)(kb1 + 32);
    // prefetch next key tile of K (CDNA5 global_prefetch_b8)
    if (kt + 1 < SEQ / 32) {
      __builtin_prefetch(kb0 + 32 * DKH, 0, 3);
      __builtin_prefetch(kb1 + 32 * DKH, 0, 3);
    }
    v8f zero = (v8f){};
    v8f s0 = wmma_f16(aq0, bl0, zero); s0 = wmma_f16(aq1, bh0, s0);
    v8f s1 = wmma_f16(aq0, bl1, zero); s1 = wmma_f16(aq1, bh1, s1);

    // ---- scale, soft mask (-1.0), online softmax over 16-lane row groups ----
#pragma unroll
    for (int r = 0; r < 8; ++r) {
      const int qrow = qrow0 + r + hi8;
      float x0 = s0[r] * 0.125f;                 // 1/sqrt(64)
      float x1 = s1[r] * 0.125f;
      if (mask[qrow * SEQ + key0 + n] == 0)       x0 = -1.0f;
      if (mask[qrow * SEQ + key0 + 16 + n] == 0)  x1 = -1.0f;
      float loc = fmaxf(x0, x1);
      loc = fmaxf(loc, __shfl_xor(loc, 1, 32));
      loc = fmaxf(loc, __shfl_xor(loc, 2, 32));
      loc = fmaxf(loc, __shfl_xor(loc, 4, 32));
      loc = fmaxf(loc, __shfl_xor(loc, 8, 32));
      const float mnew = fmaxf(mrow[r], loc);
      const float fac  = __expf(mrow[r] - mnew);
      const float p0   = __expf(x0 - mnew);
      const float p1   = __expf(x1 - mnew);
      float ps = p0 + p1;
      ps += __shfl_xor(ps, 1, 32);
      ps += __shfl_xor(ps, 2, 32);
      ps += __shfl_xor(ps, 4, 32);
      ps += __shfl_xor(ps, 8, 32);
      lrow[r] = lrow[r] * fac + ps;
      mrow[r] = mnew;
#pragma unroll
      for (int t = 0; t < 4; ++t) acc[t][r] *= fac;
      // P (D layout) -> LDS row-major 16x32
      myl[(r + hi8) * 32 + n]      = (_Float16)p0;
      myl[(r + hi8) * 32 + 16 + n] = (_Float16)p1;
    }
    __builtin_amdgcn_wave_barrier();            // LDS in-order within wave
    // ---- reload P in A-matrix layout ----
    v8h c0 = *(const v8h*)(myl + (lane & 15) * 32 + kc);
    v8h c1 = *(const v8h*)(myl + (lane & 15) * 32 + kc + 8);
    v16h pa;
#pragma unroll
    for (int i = 0; i < 8; ++i) { pa[i] = c0[i]; pa[8 + i] = c1[i]; }
    __builtin_amdgcn_wave_barrier();
    // ---- acc += P @ V (V transposed [DK][S] gives native B layout) ----
#pragma unroll
    for (int t = 0; t < 4; ++t) {
      const _Float16* vb = vbaseH + (size_t)(t * 16 + n) * SEQ + key0 + khalf16;
      v16h bv = *(const v16h*)vb;
      if (kt + 1 < SEQ / 32) __builtin_prefetch(vb + 32, 0, 3);
      acc[t] = wmma_f16(pa, bv, acc[t]);
    }
  }

  // ---- normalize and write context [B,S,D] (row-major for final GEMM A) ----
#pragma unroll
  for (int t = 0; t < 4; ++t) {
#pragma unroll
    for (int r = 0; r < 8; ++r) {
      const int sq = qrow0 + r + hi8;
      const float val = acc[t][r] / lrow[r];
      Ctx[((size_t)b * SEQ + sq) * DMODEL + h * 64 + t * 16 + n] = (_Float16)val;
    }
  }
}

// ---------------- Kernel 3: output projection (fp32 out) ----------------
// 32 rows x 64 cols per wave; pure f16 operand loads, no inner-loop cvt.
__global__ __launch_bounds__(256)
void out_proj(const _Float16* __restrict__ Ctx, const _Float16* __restrict__ woh,
              const float* __restrict__ bo, float* __restrict__ out) {
  const int lane = threadIdx.x & 31;
  const int wid  = threadIdx.x >> 5;
  const int task = blockIdx.x * 8 + wid;        // 2048 wave-tasks
  const int mtile = task >> 4;                  // 0..127
  const int ncb   = (task & 15) * 64;

  const int kc      = (lane >> 4) * 16;
  const int khalf16 = (lane >> 4) * 16;
  const int hi8     = (lane >> 4) * 8;
  const int gm0     = mtile * 32 + (lane & 15);
  const _Float16* arow0 = Ctx + (size_t)gm0 * DMODEL;
  const _Float16* arow1 = arow0 + (size_t)16 * DMODEL;
  const int ncol0 = ncb + (lane & 15);

  v8f acc0[4], acc1[4];
#pragma unroll
  for (int t = 0; t < 4; ++t) { acc0[t] = (v8f){}; acc1[t] = (v8f){}; }

  for (int kk = 0; kk < DMODEL; kk += 32) {
    v16h a0 = load_a_f16(arow0 + kk, kc);
    v16h a1 = load_a_f16(arow1 + kk, kc);
#pragma unroll
    for (int t = 0; t < 4; ++t) {
      v16h b = *(const v16h*)(woh + (size_t)(ncol0 + t * 16) * DMODEL + kk + khalf16);
      acc0[t] = wmma_f16(a0, b, acc0[t]);
      acc1[t] = wmma_f16(a1, b, acc1[t]);
    }
  }

#pragma unroll
  for (int t = 0; t < 4; ++t) {
    const int col = ncol0 + t * 16;
    const float bsv = bo[col];
#pragma unroll
    for (int r = 0; r < 8; ++r) {
      const int g0 = mtile * 32 + r + hi8;
      out[(size_t)g0 * DMODEL + col]        = acc0[t][r] + bsv;
      out[(size_t)(g0 + 16) * DMODEL + col] = acc1[t][r] + bsv;
    }
  }
}

extern "C" void kernel_launch(void* const* d_in, const int* in_sizes, int n_in,
                              void* d_out, int out_size, void* d_ws, size_t ws_size,
                              hipStream_t stream) {
  (void)in_sizes; (void)n_in; (void)out_size; (void)ws_size;
  const float* q    = (const float*)d_in[0];
  const float* k    = (const float*)d_in[1];
  const float* v    = (const float*)d_in[2];
  const int*   mask = (const int*)d_in[3];
  const float* wq   = (const float*)d_in[4];
  const float* bq   = (const float*)d_in[5];
  const float* wk   = (const float*)d_in[6];
  const float* bk   = (const float*)d_in[7];
  const float* wv   = (const float*)d_in[8];
  const float* bv   = (const float*)d_in[9];
  const float* wo   = (const float*)d_in[10];
  const float* bo   = (const float*)d_in[11];

  _Float16* wqh = (_Float16*)d_ws;          // [1024][1024] f16   2 MB
  _Float16* wkh = wqh + WELEMS;
  _Float16* wvh = wkh + WELEMS;
  _Float16* woh = wvh + WELEMS;
  _Float16* Qh  = woh + WELEMS;             // [B,H,S,DK]   8 MB
  _Float16* Kh  = Qh + HEADELEMS;           // [B,H,S,DK]   8 MB
  _Float16* Vt  = Kh + HEADELEMS;           // [B,H,DK,S]   8 MB
  _Float16* Ctx = Vt + HEADELEMS;           // [B,S,D]      8 MB

  convert_weights<<<dim3(512, 1, 4), 256, 0, stream>>>(wq, wk, wv, wo,
                                                       wqh, wkh, wvh, woh);
  proj_qkv<<<dim3(256, 1, 3), 256, 0, stream>>>(q, k, v, wqh, wkh, wvh,
                                                bq, bk, bv, Qh, Kh, Vt);
  attn_flash<<<512, 256, 0, stream>>>(Qh, Kh, Vt, mask, Ctx);
  out_proj<<<256, 256, 0, stream>>>(Ctx, woh, bo, (float*)d_out);
}
